// quantum_attention_40218073760356
// MI455X (gfx1250) — matrix operand order repaired and checked by reading, not compiled, not run on hardware
//
#include <hip/hip_runtime.h>
#include <hip/hip_bf16.h>
#include <math.h>

#define D_MODEL 1024
#define NUM_HEADS 16
#define D_KK 64
#define SEQ 2048
#define BATCH 2
#define ROWS (BATCH * SEQ)        // 4096
#define BHN (BATCH * NUM_HEADS)   // 32

typedef __attribute__((ext_vector_type(16))) __bf16 v16bf;
typedef __attribute__((ext_vector_type(8)))  __bf16 v8bf;
typedef __attribute__((ext_vector_type(8)))  float  v8f;

union Frag16 { v16bf v; v8bf h[2]; };

static __device__ __forceinline__ v8bf ld8(const __bf16* p) {
  return *reinterpret_cast<const v8bf*>(p);
}

// ---------------- fp32 -> bf16 convert ----------------
__global__ void convert_bf16(const float* __restrict__ in, __bf16* __restrict__ out, int n) {
  int i = blockIdx.x * blockDim.x + threadIdx.x;
  if (i < n) out[i] = (__bf16)in[i];
}

// ---------------- transpose 1024x1024 fp32 -> bf16 (Wt[j][k] = W[k][j]) ----------------
__global__ void transpose_bf16(const float* __restrict__ in, __bf16* __restrict__ out) {
  int idx = blockIdx.x * blockDim.x + threadIdx.x;      // 1048576
  int k = idx >> 10, j = idx & 1023;
  out[(size_t)j * D_MODEL + k] = (__bf16)in[idx];
}

// ---------------- Wo_eff[d][j] = sum_m (sum_h ent[h][m]) * Wo[m*64+d][j] ----------------
__global__ void wo_eff_kernel(const float* __restrict__ Wo, const float* __restrict__ ent,
                              float* __restrict__ woe) {
  __shared__ float c[16];
  if (threadIdx.x < 16) {
    float s = 0.f;
    for (int h = 0; h < 16; ++h) s += ent[h * 16 + threadIdx.x];
    c[threadIdx.x] = s;
  }
  __syncthreads();
  int idx = blockIdx.x * blockDim.x + threadIdx.x;      // 65536
  int d = idx >> 10, j = idx & 1023;
  float s = 0.f;
  for (int m = 0; m < 16; ++m) s += c[m] * Wo[(size_t)(m * 64 + d) * D_MODEL + j];
  woe[idx] = s;
}

// ---------------- projection GEMM: C[4096,1024] = X @ W ----------------
// Wave tile: 32 (M) x 64 (N); K loop in steps of 64 with an explicit ping-pong
// (buffer sets A/B, compile-time indices only -> no scratch demotion), so the
// next half-step's 12 loads are in flight during each 8-WMMA burst.
// vlayout==0: store bf16 as [B,H,S,dk];  vlayout==1: store bf16 as [B,H,dk,S].
__global__ __launch_bounds__(256, 1) void gemm_proj(const __bf16* __restrict__ X,
                                                    const __bf16* __restrict__ Wt,
                                                    __bf16* __restrict__ out, int vlayout) {
  const int tid  = threadIdx.x;
  const int lane = tid & 31;
  const int hlf  = lane >> 4;
  const int ln16 = lane & 15;
  const int wave = blockIdx.x * (blockDim.x >> 5) + (tid >> 5);
  const int mt = wave >> 4;        // 0..127
  const int nt = wave & 15;        // 0..15
  const int m0 = mt * 32;
  const int n0 = nt * 64;

  v8f acc[2][4];
  for (int mi = 0; mi < 2; ++mi)
    for (int t = 0; t < 4; ++t)
      for (int i = 0; i < 8; ++i) acc[mi][t][i] = 0.f;

  const __bf16* arow0 = X + (size_t)(m0 + ln16) * D_MODEL;
  const __bf16* arow1 = arow0 + (size_t)16 * D_MODEL;

  Frag16 aA[2], bA[4], aB[2], bB[4];
  auto loadStep = [&](int k0, Frag16 (&aF)[2], Frag16 (&bF)[4]) {
    aF[0].h[0] = ld8(arow0 + k0 + hlf * 8);        // e=0..7  -> K = h*8+e
    aF[0].h[1] = ld8(arow0 + k0 + 16 + hlf * 8);   // e=8..15 -> K = 16+h*8+e
    aF[1].h[0] = ld8(arow1 + k0 + hlf * 8);
    aF[1].h[1] = ld8(arow1 + k0 + 16 + hlf * 8);
    for (int t = 0; t < 4; ++t) {
      const __bf16* brow = Wt + (size_t)(n0 + t * 16 + ln16) * D_MODEL + k0 + hlf * 8;
      bF[t].h[0] = ld8(brow);                      // e=0..7 -> K = h*8+e ; e=8..15 -> K = 16+h*8+e
      bF[t].h[1] = ld8(brow + 16);
    }
  };
  auto mmaStep = [&](Frag16 (&aF)[2], Frag16 (&bF)[4]) {
    for (int mi = 0; mi < 2; ++mi)
      for (int t = 0; t < 4; ++t)
        acc[mi][t] = __builtin_amdgcn_wmma_f32_16x16x32_bf16(
            false, aF[mi].v, false, bF[t].v, (short)0, acc[mi][t], false, false);
  };

  loadStep(0, aA, bA);
#pragma unroll 1
  for (int k0 = 0; k0 < D_MODEL; k0 += 64) {
    loadStep(k0 + 32, aB, bB);                     // prefetch half-step B
    mmaStep(aA, bA);
    if (k0 + 64 < D_MODEL) loadStep(k0 + 64, aA, bA);
    mmaStep(aB, bB);
  }

  for (int mi = 0; mi < 2; ++mi)
    for (int t = 0; t < 4; ++t)
      for (int v = 0; v < 8; ++v) {
        int r = m0 + mi * 16 + v + 8 * hlf;        // C: VGPR v -> M = v + 8*half
        int j = n0 + t * 16 + ln16;
        int b = r >> 11, s = r & (SEQ - 1);
        int hh = j >> 6, d = j & 63;
        size_t idx = vlayout
          ? ((size_t)(b * NUM_HEADS + hh) * D_KK + d) * SEQ + s
          : ((size_t)(b * NUM_HEADS + hh) * SEQ + s) * D_KK + d;
        out[idx] = (__bf16)acc[mi][t][v];
      }
}

// ---------------- fused attention: per (b,h), streaming max-free softmax ----------------
// Block = 8 waves; wave owns 16 query rows. Per 32-key half-step: V-fragment
// loads first (overlap score WMMAs + transcendentals), K fragments in an
// explicit ping-pong across half-steps, probs staged via LDS into A-fragments.
__global__ __launch_bounds__(256, 1) void attention_kernel(
    const __bf16* __restrict__ Qb, const __bf16* __restrict__ Kb,
    const __bf16* __restrict__ Vt, const float* __restrict__ phase_scale,
    float* __restrict__ att) {
  __shared__ __align__(16) __bf16 pbuf[8][16][32];
  const int tid  = threadIdx.x;
  const int lane = tid & 31;
  const int hlf  = lane >> 4;
  const int ln16 = lane & 15;
  const int wave = tid >> 5;
  const int bh = blockIdx.x >> 4;            // 0..31
  const int qblk = blockIdx.x & 15;
  const int q0 = qblk * 128 + wave * 16;
  const float scale = 0.125f * phase_scale[bh & (NUM_HEADS - 1)];  // 1/sqrt(64)/T * ps[h]

  // Q A-fragments for the two 32-wide dk chunks
  const __bf16* qbase = Qb + ((size_t)bh * SEQ + q0) * D_KK;
  Frag16 qf[2];
  for (int ks = 0; ks < 2; ++ks) {
    const __bf16* ap = qbase + ln16 * D_KK + ks * 32 + hlf * 16;
    qf[ks].h[0] = ld8(ap);
    qf[ks].h[1] = ld8(ap + 8);
  }

  v8f accN[4];
  for (int t = 0; t < 4; ++t)
    for (int i = 0; i < 8; ++i) accN[t][i] = 0.f;
  float den[8];
  for (int v = 0; v < 8; ++v) den[v] = 0.f;

  Frag16 kA[2][2], kB[2][2];                 // [kt][ks]
  auto loadK = [&](int k0, Frag16 (&kf)[2][2]) {
    for (int kt = 0; kt < 2; ++kt) {
      const int kk = k0 + kt * 16 + ln16;    // this lane's key (B-matrix column n)
      const __bf16* kp = Kb + ((size_t)bh * SEQ + kk) * D_KK + hlf * 16;
      kf[kt][0].h[0] = ld8(kp);              // ks=0: K = h*16+e
      kf[kt][0].h[1] = ld8(kp + 8);
      kf[kt][1].h[0] = ld8(kp + 32);         // ks=1: +32 elements
      kf[kt][1].h[1] = ld8(kp + 40);
    }
  };
  auto step = [&](int k0, Frag16 (&kf)[2][2], Frag16 (&kn)[2][2], bool prefetch) {
    // V B-fragments for this step: independent of scores, issue first
    Frag16 vf[4];
    for (int dt = 0; dt < 4; ++dt) {
      const __bf16* vp = Vt + ((size_t)bh * D_KK + dt * 16 + ln16) * SEQ + k0 + hlf * 8;
      vf[dt].h[0] = ld8(vp);
      vf[dt].h[1] = ld8(vp + 16);
    }
    if (prefetch) loadK(k0 + 32, kn);        // next half-step's K in flight

    for (int kt = 0; kt < 2; ++kt) {
      v8f sacc;
      for (int i = 0; i < 8; ++i) sacc[i] = 0.f;
      sacc = __builtin_amdgcn_wmma_f32_16x16x32_bf16(
          false, qf[0].v, false, kf[kt][0].v, (short)0, sacc, false, false);
      sacc = __builtin_amdgcn_wmma_f32_16x16x32_bf16(
          false, qf[1].v, false, kf[kt][1].v, (short)0, sacc, false, false);
      for (int v = 0; v < 8; ++v) {
        float sv = sacc[v] * scale;
        float wv = __cosf(sv) + 0.1f * __cosf(2.0f * sv);  // bounded in [-1.1, 1.1]
        float ev = __expf(wv);                              // safe without max-subtraction
        den[v] += ev;
        pbuf[wave][v + 8 * hlf][kt * 16 + ln16] = (__bf16)ev;
      }
    }
    // Reload probs as A-fragment (16 x 32 keys): e=0..7 -> K=h*8+e ; e=8..15 -> K=16+h*8+e
    Frag16 p;
    p.h[0] = ld8(&pbuf[wave][ln16][hlf * 8]);
    p.h[1] = ld8(&pbuf[wave][ln16][16 + hlf * 8]);
    for (int dt = 0; dt < 4; ++dt)
      accN[dt] = __builtin_amdgcn_wmma_f32_16x16x32_bf16(
          false, p.v, false, vf[dt].v, (short)0, accN[dt], false, false);
  };

  loadK(0, kA);
#pragma unroll 1
  for (int k0 = 0; k0 < SEQ; k0 += 64) {
    step(k0, kA, kB, true);
    step(k0 + 32, kB, kA, k0 + 64 < SEQ);
  }

  // row-sum reduce den over 16 lanes of each half (rows live within one half)
  float rden[8];
  for (int v = 0; v < 8; ++v) {
    float s = den[v];
    s += __shfl_xor(s, 1, 32);
    s += __shfl_xor(s, 2, 32);
    s += __shfl_xor(s, 4, 32);
    s += __shfl_xor(s, 8, 32);
    rden[v] = 1.0f / s;
  }
  for (int dt = 0; dt < 4; ++dt)
    for (int v = 0; v < 8; ++v) {
      int q = q0 + v + 8 * hlf;
      int d = dt * 16 + ln16;
      att[((size_t)bh * SEQ + q) * D_KK + d] = accN[dt][v] * rden[v];
    }
}

// ---------------- A[b,s,d] = sum_h att[b,h,s,d] ----------------
__global__ void head_sum(const float* __restrict__ att, float* __restrict__ A) {
  int idx = blockIdx.x * blockDim.x + threadIdx.x;       // ROWS*64
  if (idx >= ROWS * D_KK) return;
  int d = idx & 63;
  int rs = idx >> 6;
  int b = rs >> 11, s = rs & (SEQ - 1);
  float acc = 0.f;
  for (int h = 0; h < NUM_HEADS; ++h)
    acc += att[(((size_t)(b * NUM_HEADS + h)) * SEQ + s) * D_KK + d];
  A[idx] = acc;
}

// ---------------- out = LN(A @ Wo_eff + bo + x) ----------------
__global__ void out_ln(const float* __restrict__ A, const float* __restrict__ Woe,
                       const float* __restrict__ bo, const float* __restrict__ x,
                       const float* __restrict__ gamma, const float* __restrict__ beta,
                       float* __restrict__ out) {
  __shared__ float sA[64];
  __shared__ float red[256];
  const int row = blockIdx.x;
  const int tid = threadIdx.x;
  if (tid < 64) sA[tid] = A[(size_t)row * D_KK + tid];
  __syncthreads();
  float rv[4];
  float lsum = 0.f, lsq = 0.f;
  for (int c = 0; c < 4; ++c) {
    int j = tid + 256 * c;
    float acc = bo[j];
    for (int d = 0; d < 64; ++d) acc += sA[d] * Woe[(size_t)d * D_MODEL + j];
    float r = acc + x[(size_t)row * D_MODEL + j];
    rv[c] = r;
    lsum += r;
    lsq += r * r;
  }
  red[tid] = lsum; __syncthreads();
  for (int s = 128; s > 0; s >>= 1) { if (tid < s) red[tid] += red[tid + s]; __syncthreads(); }
  float mu = red[0] * (1.0f / D_MODEL);
  __syncthreads();
  red[tid] = lsq; __syncthreads();
  for (int s = 128; s > 0; s >>= 1) { if (tid < s) red[tid] += red[tid + s]; __syncthreads(); }
  float var = red[0] * (1.0f / D_MODEL) - mu * mu;
  float inv = rsqrtf(var + 1e-6f);
  for (int c = 0; c < 4; ++c) {
    int j = tid + 256 * c;
    out[(size_t)row * D_MODEL + j] = gamma[j] * (rv[c] - mu) * inv + beta[j];
  }
}

extern "C" void kernel_launch(void* const* d_in, const int* in_sizes, int n_in,
                              void* d_out, int out_size, void* d_ws, size_t ws_size,
                              hipStream_t stream) {
  (void)in_sizes; (void)n_in; (void)out_size; (void)ws_size;
  const float* x     = (const float*)d_in[0];
  const float* Wq    = (const float*)d_in[1];
  const float* Wk    = (const float*)d_in[2];
  const float* Wv    = (const float*)d_in[3];
  const float* Wo    = (const float*)d_in[4];
  const float* bo    = (const float*)d_in[5];
  const float* ps    = (const float*)d_in[6];
  const float* ent   = (const float*)d_in[7];
  const float* gamma = (const float*)d_in[8];
  const float* beta  = (const float*)d_in[9];
  float* out = (float*)d_out;

  char* w = (char*)d_ws;
  size_t off = 0;
  auto take = [&](size_t bytes) -> void* {
    void* p = w + off;
    off += (bytes + 255) & ~(size_t)255;
    return p;
  };
  __bf16* xb   = (__bf16*)take((size_t)ROWS * D_MODEL * 2);          // 8 MiB
  __bf16* wqt  = (__bf16*)take((size_t)D_MODEL * D_MODEL * 2);       // 2 MiB
  __bf16* wkt  = (__bf16*)take((size_t)D_MODEL * D_MODEL * 2);
  __bf16* wvt  = (__bf16*)take((size_t)D_MODEL * D_MODEL * 2);
  __bf16* qbuf = (__bf16*)take((size_t)BHN * SEQ * D_KK * 2);        // 8 MiB
  __bf16* kbuf = (__bf16*)take((size_t)BHN * SEQ * D_KK * 2);
  __bf16* vbuf = (__bf16*)take((size_t)BHN * SEQ * D_KK * 2);
  float*  attw = (float*)take((size_t)BHN * SEQ * D_KK * 4);         // 16 MiB
  float*  Abuf = (float*)take((size_t)ROWS * D_KK * 4);              // 1 MiB
  float*  woe  = (float*)take((size_t)D_KK * D_MODEL * 4);           // 256 KiB

  convert_bf16<<<(ROWS * D_MODEL) / 256, 256, 0, stream>>>(x, xb, ROWS * D_MODEL);
  transpose_bf16<<<(D_MODEL * D_MODEL) / 256, 256, 0, stream>>>(Wq, wqt);
  transpose_bf16<<<(D_MODEL * D_MODEL) / 256, 256, 0, stream>>>(Wk, wkt);
  transpose_bf16<<<(D_MODEL * D_MODEL) / 256, 256, 0, stream>>>(Wv, wvt);
  wo_eff_kernel<<<(D_KK * D_MODEL) / 256, 256, 0, stream>>>(Wo, ent, woe);

  // 2048 wave-tiles (128 M-tiles x 16 N-tiles), 8 waves/block
  gemm_proj<<<256, 256, 0, stream>>>(xb, wqt, qbuf, 0);
  gemm_proj<<<256, 256, 0, stream>>>(xb, wkt, kbuf, 0);
  gemm_proj<<<256, 256, 0, stream>>>(xb, wvt, vbuf, 1);

  attention_kernel<<<BHN * 16, 256, 0, stream>>>(qbuf, kbuf, vbuf, ps, attw);
  head_sum<<<(ROWS * D_KK) / 256, 256, 0, stream>>>(attw, Abuf);
  out_ln<<<ROWS, 256, 0, stream>>>(Abuf, woe, bo, x, gamma, beta, out);
}
